// CSAttention_50027779063818
// MI455X (gfx1250) — matrix operand order repaired and checked by reading, not compiled, not run on hardware
//
#include <hip/hip_runtime.h>
#include <hip/hip_bf16.h>
#include <cstdint>

typedef __attribute__((ext_vector_type(16))) _Float16 v16h;
typedef __attribute__((ext_vector_type(8)))  float    v8f;

// ---------------------------------------------------------------------------
// WMMA GEMM: Y(MxN) = A(MxK) @ W(KxN) + bias, f32 accumulate.
// Fragment-major LDS layout: each lane's WMMA operand is one contiguous v16h
// (32 bytes -> 2x ds_load_b128) instead of 16 strided ds_load_u16.
//   Asv[row*2 + h] : A-fragment halves for lane (r=row%16, h) of wave row/16
//                    element e -> A[row][k0 + (e<8 ? 8h+e : 16+8h+(e-8))]
//   Bsv[col*2 + h] : B-fragment halves, element e -> W[k0 + (e<8 ? 8h+e : 16+8h+(e-8))][col]
// EPI: 0 store, 2 store+blend-init, 3 store+blend-acc, 4 accumulate-into-Y
// Block = 128 threads (4 waves) -> 64x64 tile, K step 32.
// ---------------------------------------------------------------------------
template<typename AT, int EPI>
__global__ void __launch_bounds__(128)
gemm_wmma(const AT* __restrict__ A, const float* __restrict__ W,
          const float* __restrict__ bias, float* __restrict__ Yf,
          int N, int K,
          const float* __restrict__ wts, int wsel, float* __restrict__ blend)
{
    __shared__ v16h Asv[128];   // 64 rows x 2 halves
    __shared__ v16h Bsv[128];   // 64 cols x 2 halves

    const int tid  = threadIdx.x;
    const int wave = tid >> 5;
    const int lane = tid & 31;
    const int row0 = blockIdx.x * 64;
    const int col0 = blockIdx.y * 64;
    const int lr   = lane & 15;   // row/col within 16
    const int lh   = lane >> 4;   // lane half

    v8f acc[4];
#pragma unroll
    for (int ct = 0; ct < 4; ++ct)
#pragma unroll
        for (int e = 0; e < 8; ++e) acc[ct][e] = 0.0f;

    for (int k0 = 0; k0 < K; k0 += 32) {
        // ---- stage A: thread t -> (row = t>>1, h = t&1); two contiguous
        //      8-element runs of the global row, one vector LDS store.
        {
            const int row = tid >> 1, h = tid & 1;
            const AT* ap = A + (size_t)(row0 + row) * K + k0;
            v16h fr;
#pragma unroll
            for (int e = 0; e < 8; ++e) fr[e]     = (_Float16)ap[8 * h + e];
#pragma unroll
            for (int e = 0; e < 8; ++e) fr[8 + e] = (_Float16)ap[16 + 8 * h + e];
            Asv[row * 2 + h] = fr;
        }
        // ---- stage B: thread t reads 16 row-contiguous W elements
        //      (kk = t>>2 fixed, 16 consecutive cols), scatters b16 stores.
        {
            const int kk = tid >> 2;
            const int cbase = (tid & 3) * 16;
            const int h = (kk >> 3) & 1, e = (kk & 7) + ((kk >> 4) << 3);
            const float* wp = W + (size_t)(k0 + kk) * N + col0 + cbase;
            _Float16* bp = (_Float16*)Bsv;
#pragma unroll
            for (int i = 0; i < 16; ++i)
                bp[((cbase + i) * 2 + h) * 16 + e] = (_Float16)wp[i];
        }
        __syncthreads();

        const v16h af = Asv[(wave * 16 + lr) * 2 + lh];
#pragma unroll
        for (int ct = 0; ct < 4; ++ct) {
            const v16h bf = Bsv[(ct * 16 + lr) * 2 + lh];
            acc[ct] = __builtin_amdgcn_wmma_f32_16x16x32_f16(
                false, af, false, bf, (short)0, acc[ct], false, false);
        }
        __syncthreads();
    }

    // epilogue (C/D layout: lane%16 = col, vgpr g -> row g + 8*(lane/16))
#pragma unroll
    for (int ct = 0; ct < 4; ++ct) {
        int gc = col0 + ct * 16 + lr;
        float bv = bias ? bias[gc] : 0.0f;
#pragma unroll
        for (int g = 0; g < 8; ++g) {
            int gr = row0 + wave * 16 + g + 8 * lh;
            size_t o = (size_t)gr * N + gc;
            float v = acc[ct][g] + bv;
            if (EPI == 0) {
                Yf[o] = v;
            } else if (EPI == 2) {
                Yf[o] = v;
                blend[o] = wts[(size_t)gr * 3 + wsel] * v;
            } else if (EPI == 3) {
                Yf[o] = v;
                blend[o] += wts[(size_t)gr * 3 + wsel] * v;
            } else { // 4
                Yf[o] += v;
            }
        }
    }
}

// ---------------------------------------------------------------------------
// Window attention for one scale.  One block per (b, p1, p2) window.
// q/k/v staged f16 in LDS; scores use v16h row loads, ctx reads 8 contiguous
// channels per thread so LDS traffic is b128s, not scalar u16 gathers.
// ---------------------------------------------------------------------------
template<int WS>
__global__ void __launch_bounds__(128)
win_attn(const float* __restrict__ Q, const float* __restrict__ K,
         const float* __restrict__ V, _Float16* __restrict__ CTX)
{
    constexpr int WIN = WS * WS;
    constexpr int NP  = 64 / WS;
    __shared__ _Float16 qs[WIN][256];
    __shared__ _Float16 ks[WIN][256];
    __shared__ _Float16 vs[WIN][256];
    __shared__ float    sc[WIN][WIN + 1];

    const int blk = blockIdx.x;
    const int b  = blk / (NP * NP);
    const int p1 = (blk / NP) % NP;
    const int p2 = blk % NP;
    const int tid = threadIdx.x;

    for (int idx = tid; idx < WIN * 256; idx += 128) {
        int t = idx >> 8, c = idx & 255;
        int r = t / WS, cc = t % WS;
        size_t g = (((size_t)(b * 64 + p1 * WS + r)) * 64 + (p2 * WS + cc)) * 256 + c;
        qs[t][c] = (_Float16)Q[g];
        ks[t][c] = (_Float16)K[g];
        vs[t][c] = (_Float16)V[g];
    }
    __syncthreads();

    const float rs = 0.17677669529663688f; // 1/sqrt(32)
    for (int hh = 0; hh < 8; ++hh) {
        const int d0 = hh * 32;
        // scores: vector loads of 16-half chunks
        for (int idx = tid; idx < WIN * WIN; idx += 128) {
            int i = idx / WIN, j = idx % WIN;
            const v16h qa = *(const v16h*)&qs[i][d0];
            const v16h qb = *(const v16h*)&qs[i][d0 + 16];
            const v16h ka = *(const v16h*)&ks[j][d0];
            const v16h kb = *(const v16h*)&ks[j][d0 + 16];
            float s = 0.0f;
#pragma unroll
            for (int e = 0; e < 16; ++e)
                s += (float)qa[e] * (float)ka[e] + (float)qb[e] * (float)kb[e];
            sc[i][j] = s * rs;
        }
        __syncthreads();
        // softmax rows
        for (int i = tid; i < WIN; i += 128) {
            float m = -1e30f;
            for (int j = 0; j < WIN; ++j) m = fmaxf(m, sc[i][j]);
            float sum = 0.0f;
            for (int j = 0; j < WIN; ++j) { float e = __expf(sc[i][j] - m); sc[i][j] = e; sum += e; }
            float inv = 1.0f / sum;
            for (int j = 0; j < WIN; ++j) sc[i][j] *= inv;
        }
        __syncthreads();
        // ctx: thread owns (i, 8 contiguous channels) -> contiguous v reads
        for (int idx = tid; idx < WIN * 4; idx += 128) {
            int i = idx >> 2, dc = (idx & 3) * 8;
            float a0 = 0, a1 = 0, a2 = 0, a3 = 0, a4 = 0, a5 = 0, a6 = 0, a7 = 0;
            for (int j = 0; j < WIN; ++j) {
                float p = sc[i][j];
                const _Float16* vp = &vs[j][d0 + dc];
                a0 += p * (float)vp[0]; a1 += p * (float)vp[1];
                a2 += p * (float)vp[2]; a3 += p * (float)vp[3];
                a4 += p * (float)vp[4]; a5 += p * (float)vp[5];
                a6 += p * (float)vp[6]; a7 += p * (float)vp[7];
            }
            int r = i / WS, cc = i % WS;
            size_t g = (((size_t)(b * 64 + p1 * WS + r)) * 64 + (p2 * WS + cc)) * 256 + d0 + dc;
            CTX[g + 0] = (_Float16)a0; CTX[g + 1] = (_Float16)a1;
            CTX[g + 2] = (_Float16)a2; CTX[g + 3] = (_Float16)a3;
            CTX[g + 4] = (_Float16)a4; CTX[g + 5] = (_Float16)a5;
            CTX[g + 6] = (_Float16)a6; CTX[g + 7] = (_Float16)a7;
        }
        __syncthreads();
    }
}

// ---------------------------------------------------------------------------
// Edge conv: |3x3 conv over C=256 channels| -> (B,H,W) scalar map
// ---------------------------------------------------------------------------
__global__ void edge_conv(const float* __restrict__ X, const float* __restrict__ EW,
                          float* __restrict__ E)
{
    const int b = blockIdx.x >> 6;
    const int y = blockIdx.x & 63;
    const int x = threadIdx.x;
    float s = 0.0f;
    for (int ky = 0; ky < 3; ++ky) {
        int yy = y + ky - 1;
        if (yy < 0 || yy > 63) continue;
        for (int kx = 0; kx < 3; ++kx) {
            int xx = x + kx - 1;
            if (xx < 0 || xx > 63) continue;
            const float* xp = X + (((size_t)(b * 64 + yy)) * 64 + xx) * 256;
            float acc = 0.0f;
            for (int c = 0; c < 256; ++c)
                acc += xp[c] * EW[c * 9 + ky * 3 + kx];
            s += acc;
        }
    }
    E[((size_t)(b * 64 + y)) * 64 + x] = fabsf(s);
}

// 3x3 avg of edge map -> 1->16->3 MLP -> softmax  => per-pixel scale weights
__global__ void selector(const float* __restrict__ E, const float* __restrict__ w1,
                         const float* __restrict__ b1, const float* __restrict__ w2,
                         const float* __restrict__ b2, float* __restrict__ WTS)
{
    int idx = blockIdx.x * 256 + threadIdx.x;
    if (idx >= 16 * 64 * 64) return;
    int b = idx >> 12, y = (idx >> 6) & 63, x = idx & 63;
    float s = 0.0f;
    for (int dy = -1; dy <= 1; ++dy)
        for (int dx = -1; dx <= 1; ++dx) {
            int yy = y + dy, xx = x + dx;
            if (yy >= 0 && yy < 64 && xx >= 0 && xx < 64)
                s += E[((size_t)(b * 64 + yy)) * 64 + xx];
        }
    float e = s / 9.0f;
    float l0 = b2[0], l1 = b2[1], l2 = b2[2];
    for (int j = 0; j < 16; ++j) {
        float hv = fmaxf(e * w1[j] + b1[j], 0.0f);
        l0 += hv * w2[j * 3 + 0];
        l1 += hv * w2[j * 3 + 1];
        l2 += hv * w2[j * 3 + 2];
    }
    float m = fmaxf(l0, fmaxf(l1, l2));
    float e0 = __expf(l0 - m), e1 = __expf(l1 - m), e2 = __expf(l2 - m);
    float inv = 1.0f / (e0 + e1 + e2);
    WTS[(size_t)idx * 3 + 0] = e0 * inv;
    WTS[(size_t)idx * 3 + 1] = e1 * inv;
    WTS[(size_t)idx * 3 + 2] = e2 * inv;
}

// ---------------------------------------------------------------------------
// DlightConv: mean over 16 window tokens -> C->16 linear -> softmax ->
// weighted sum of tokens.  One block (256 thr) per (b,p1,p2) window.
// ---------------------------------------------------------------------------
__global__ void __launch_bounds__(256)
dlight(const float* __restrict__ XWIN, const float* __restrict__ dlw,
       const float* __restrict__ dlb, float* __restrict__ XDL)
{
    __shared__ float mean[256];
    __shared__ float pr[16];
    const int blk = blockIdx.x;
    const int b = blk >> 8, p1 = (blk >> 4) & 15, p2 = blk & 15;
    const int c = threadIdx.x;
    float xv[16];
    float m = 0.0f;
#pragma unroll
    for (int t = 0; t < 16; ++t) {
        int r = t >> 2, cc = t & 3;
        xv[t] = XWIN[(((size_t)(b * 64 + p1 * 4 + r)) * 64 + (p2 * 4 + cc)) * 256 + c];
        m += xv[t];
    }
    mean[c] = m * (1.0f / 16.0f);
    __syncthreads();
    if (c < 16) {
        float lg = dlb[c];
        for (int k = 0; k < 256; ++k) lg += mean[k] * dlw[k * 16 + c];
        pr[c] = lg;
    }
    __syncthreads();
    if (c == 0) {
        float mx = pr[0];
        for (int j = 1; j < 16; ++j) mx = fmaxf(mx, pr[j]);
        float sm = 0.0f;
        for (int j = 0; j < 16; ++j) { pr[j] = __expf(pr[j] - mx); sm += pr[j]; }
        float inv = 1.0f / sm;
        for (int j = 0; j < 16; ++j) pr[j] *= inv;
    }
    __syncthreads();
    float o = 0.0f;
#pragma unroll
    for (int t = 0; t < 16; ++t) o += xv[t] * pr[t];
    XDL[(size_t)blk * 256 + c] = o;
}

__device__ inline void softmax16(float* a)
{
    float m = a[0];
    for (int i = 1; i < 16; ++i) m = fmaxf(m, a[i]);
    float s = 0.0f;
    for (int i = 0; i < 16; ++i) { a[i] = __expf(a[i] - m); s += a[i]; }
    float inv = 1.0f / s;
    for (int i = 0; i < 16; ++i) a[i] *= inv;
}

// ---------------------------------------------------------------------------
// Axial attention with Gaussian positional bias on the 16x16 pooled grid.
// One block (256 thr = channels) per (b, r, c) position.
// ---------------------------------------------------------------------------
__global__ void __launch_bounds__(256)
axial(const float* __restrict__ Qd, const float* __restrict__ Kd,
      const float* __restrict__ Vd, const float* __restrict__ gshift,
      const float* __restrict__ gbias, float* __restrict__ XG)
{
    __shared__ float qv[256];
    __shared__ float ax[16], ay[16];
    const int blk = blockIdx.x;
    const int b = blk >> 8, r = (blk >> 4) & 15, c = blk & 15;
    const int d = threadIdx.x;
    qv[d] = Qd[(size_t)blk * 256 + d];
    __syncthreads();
    const float gs = gshift[0], gb = gbias[0];
    if (d < 16) { // row attention: keys are columns k of row r, bias -(gs*(c-k)^2+gb)
        const float* kp = Kd + ((size_t)((b * 16 + r) * 16 + d)) * 256;
        float s = 0.0f;
        for (int e = 0; e < 256; ++e) s += qv[e] * kp[e];
        float dist = (float)(c - d);
        ax[d] = s - (gs * dist * dist + gb);
    } else if (d < 32) { // column attention: keys are rows k of column c, bias by (r-k)
        int k = d - 16;
        const float* kp = Kd + ((size_t)((b * 16 + k) * 16 + c)) * 256;
        float s = 0.0f;
        for (int e = 0; e < 256; ++e) s += qv[e] * kp[e];
        float dist = (float)(r - k);
        ay[k] = s - (gs * dist * dist + gb);
    }
    __syncthreads();
    if (d == 0)  softmax16(ax);
    if (d == 32) softmax16(ay);
    __syncthreads();
    float o = 0.0f;
#pragma unroll
    for (int k = 0; k < 16; ++k) {
        o += ax[k] * Vd[((size_t)((b * 16 + r) * 16 + k)) * 256 + d];
        o += ay[k] * Vd[((size_t)((b * 16 + k) * 16 + c)) * 256 + d];
    }
    XG[(size_t)blk * 256 + d] = o;
}

// bilinear upsample 16x16 -> 64x64, align_corners=True, output f16 for GEMM A
__global__ void __launch_bounds__(256)
upsample(const float* __restrict__ XG, _Float16* __restrict__ OUT)
{
    const int by = blockIdx.x;
    const int b = by >> 6, y = by & 63;
    const float scl = 15.0f / 63.0f;
    float py = y * scl;
    int r0 = (int)py;
    int r1 = (r0 + 1 < 15) ? r0 + 1 : 15;
    float tr = py - (float)r0;
    const int cch = threadIdx.x;
    const float* base = XG + (size_t)b * 16 * 16 * 256;
    for (int x = 0; x < 64; ++x) {
        float px = x * scl;
        int c0 = (int)px;
        int c1 = (c0 + 1 < 15) ? c0 + 1 : 15;
        float tc = px - (float)c0;
        float v00 = base[((r0 * 16) + c0) * 256 + cch];
        float v01 = base[((r0 * 16) + c1) * 256 + cch];
        float v10 = base[((r1 * 16) + c0) * 256 + cch];
        float v11 = base[((r1 * 16) + c1) * 256 + cch];
        float v = (v00 * (1.0f - tc) + v01 * tc) * (1.0f - tr)
                + (v10 * (1.0f - tc) + v11 * tc) * tr;
        OUT[(((size_t)(b * 64 + y)) * 64 + x) * 256 + cch] = (_Float16)v;
    }
}

// sq_w (256 x 512) -> transposed (512 x 256) so it can be a GEMM B operand
__global__ void transpose_sqw(const float* __restrict__ SW, float* __restrict__ SWT)
{
    int idx = blockIdx.x * 256 + threadIdx.x;
    if (idx >= 256 * 512) return;
    int o = idx / 512, k = idx % 512;
    SWT[(size_t)k * 256 + o] = SW[idx];
}

// ---------------------------------------------------------------------------
extern "C" void kernel_launch(void* const* d_in, const int* in_sizes, int n_in,
                              void* d_out, int out_size, void* d_ws, size_t ws_size,
                              hipStream_t stream)
{
    (void)in_sizes; (void)n_in; (void)out_size; (void)ws_size;
    const float* x      = (const float*)d_in[0];
    const float* attn_w = (const float*)d_in[1];
    const float* attn_b = (const float*)d_in[2];
    const float* edge_w = (const float*)d_in[3];
    const float* mlp_w1 = (const float*)d_in[4];
    const float* mlp_b1 = (const float*)d_in[5];
    const float* mlp_w2 = (const float*)d_in[6];
    const float* mlp_b2 = (const float*)d_in[7];
    const float* dl_w   = (const float*)d_in[8];
    const float* dl_b   = (const float*)d_in[9];
    const float* ax_w   = (const float*)d_in[10];
    const float* ax_b   = (const float*)d_in[11];
    const float* g_shift= (const float*)d_in[12];
    const float* g_bias = (const float*)d_in[13];
    const float* sq_w   = (const float*)d_in[14];
    const float* sq_b   = (const float*)d_in[15];

    const int B = 16, C = 256;
    const size_t M = (size_t)B * 64 * 64; // 65536 tokens
    const size_t MD = (size_t)B * 256;    // 4096 pooled tokens

    char* ws = (char*)d_ws;
    size_t off = 0;
    auto carve = [&](size_t bytes) -> void* {
        void* p = ws + off;
        off += (bytes + 255) & ~(size_t)255;
        return p;
    };
    float*    Q    = (float*)carve(M * C * 4);
    float*    Kb   = (float*)carve(M * C * 4);
    float*    V    = (float*)carve(M * C * 4);
    _Float16* CTX  = (_Float16*)carve(M * C * 2);
    float*    XWIN = (float*)carve(M * C * 4);   // ws=4 branch output (reused for Dlight)
    float*    ATTN = (float*)carve(M * C * 4);   // ws=2 / ws=8 branch outputs
    float*    BLND = (float*)carve(M * C * 4);   // blended h_map
    float*    EDGE = (float*)carve(M * 4);
    float*    WTS  = (float*)carve(M * 3 * 4);
    float*    XDL  = (float*)carve(MD * C * 4);
    float*    AXQ  = (float*)carve(MD * C * 4);
    float*    AXK  = (float*)carve(MD * C * 4);
    float*    AXV  = (float*)carve(MD * C * 4);
    float*    XG   = (float*)carve(MD * C * 4);
    _Float16* XGUP = (_Float16*)carve(M * C * 2);
    float*    SQWT = (float*)carve((size_t)512 * 256 * 4);

    float* OUT = (float*)d_out;

    const dim3 gblk(128);
    const dim3 ggrd((unsigned)(M / 64), (unsigned)(C / 64));   // 1024 x 4
    const dim3 agrd((unsigned)(MD / 64), (unsigned)(C / 64));  // 64 x 4

    // ---- adaptive window selector (needs only x) ----
    transpose_sqw<<<512, 256, 0, stream>>>(sq_w, SQWT);
    edge_conv<<<B * 64, 64, 0, stream>>>(x, edge_w, EDGE);
    selector<<<(unsigned)(M / 256), 256, 0, stream>>>(EDGE, mlp_w1, mlp_b1, mlp_w2, mlp_b2, WTS);

    // ---- multi-scale window attention, blend fused into out-projection ----
    for (int i = 0; i < 3; ++i) {
        const float* Wq = attn_w + ((size_t)i * 4 + 0) * C * C;
        const float* Wk = attn_w + ((size_t)i * 4 + 1) * C * C;
        const float* Wv = attn_w + ((size_t)i * 4 + 2) * C * C;
        const float* Wo = attn_w + ((size_t)i * 4 + 3) * C * C;
        const float* bq = attn_b + ((size_t)i * 4 + 0) * C;
        const float* bk = attn_b + ((size_t)i * 4 + 1) * C;
        const float* bv = attn_b + ((size_t)i * 4 + 2) * C;
        const float* bo = attn_b + ((size_t)i * 4 + 3) * C;

        gemm_wmma<float, 0><<<ggrd, gblk, 0, stream>>>(x, Wq, bq, Q,  C, C, nullptr, 0, nullptr);
        gemm_wmma<float, 0><<<ggrd, gblk, 0, stream>>>(x, Wk, bk, Kb, C, C, nullptr, 0, nullptr);
        gemm_wmma<float, 0><<<ggrd, gblk, 0, stream>>>(x, Wv, bv, V,  C, C, nullptr, 0, nullptr);

        if (i == 0)      win_attn<2><<<B * 32 * 32, 128, 0, stream>>>(Q, Kb, V, CTX);
        else if (i == 1) win_attn<4><<<B * 16 * 16, 128, 0, stream>>>(Q, Kb, V, CTX);
        else             win_attn<8><<<B *  8 *  8, 128, 0, stream>>>(Q, Kb, V, CTX);

        float* dst = (i == 1) ? XWIN : ATTN;
        if (i == 0)
            gemm_wmma<_Float16, 2><<<ggrd, gblk, 0, stream>>>(CTX, Wo, bo, dst, C, C, WTS, i, BLND);
        else
            gemm_wmma<_Float16, 3><<<ggrd, gblk, 0, stream>>>(CTX, Wo, bo, dst, C, C, WTS, i, BLND);
    }

    // ---- DlightConv pooling on the ws=4 branch output ----
    dlight<<<(unsigned)MD, 256, 0, stream>>>(XWIN, dl_w, dl_b, XDL);

    // ---- axial attention projections + attention + gaussian bias ----
    gemm_wmma<float, 0><<<agrd, gblk, 0, stream>>>(XDL, ax_w + 0 * C * C, ax_b + 0 * C, AXQ, C, C, nullptr, 0, nullptr);
    gemm_wmma<float, 0><<<agrd, gblk, 0, stream>>>(XDL, ax_w + 1 * C * C, ax_b + 1 * C, AXK, C, C, nullptr, 0, nullptr);
    gemm_wmma<float, 0><<<agrd, gblk, 0, stream>>>(XDL, ax_w + 2 * C * C, ax_b + 2 * C, AXV, C, C, nullptr, 0, nullptr);
    axial<<<(unsigned)MD, 256, 0, stream>>>(AXQ, AXK, AXV, g_shift, g_bias, XG);

    // ---- upsample 16x16 -> 64x64 ----
    upsample<<<B * 64, 256, 0, stream>>>(XG, XGUP);

    // ---- squeeze: out = x_g_up @ sqT[0:256] + blended @ sqT[256:512] + sq_b ----
    gemm_wmma<_Float16, 0><<<ggrd, gblk, 0, stream>>>(XGUP, SQWT,            nullptr, OUT, C, C, nullptr, 0, nullptr);
    gemm_wmma<float,    4><<<ggrd, gblk, 0, stream>>>(BLND, SQWT + 256 * 256, sq_b,   OUT, C, C, nullptr, 0, nullptr);
}
